// Twinter_Net_8796093022245
// MI455X (gfx1250) — matrix operand order repaired and checked by reading, not compiled, not run on hardware
//
#include <hip/hip_runtime.h>
#include <hip/hip_bf16.h>

// ---------------------------------------------------------------------------
// Twinter Net fused forward for gfx1250 (MI455X), bf16 WMMA path.
//   P=Q=32, R=8, H1=128, H2=64, PH=16, B=8192, PAIRS=1024, OUT=8192
// Never materialize h[B,8192] (256MB f32) to HBM: fuse the interaction GEMM
// with the 1024 per-pair MLPs per 16x16 tile. Tiles round-trip through LDS
// as bf16 via d16_hi truncating stores (bf16 == f32[31:16]). Per-pair bias
// is folded into the WMMA as a K=8 bias row (A=1.0, B=pb1), so the per-pair
// WMMA uses the inline-0 C operand and needs no splats or selects.
// ---------------------------------------------------------------------------

typedef __bf16 bf16_t;
typedef __attribute__((ext_vector_type(16))) __bf16 v16bf;
typedef __attribute__((ext_vector_type(8)))  __bf16 v8bf;
typedef __attribute__((ext_vector_type(8)))  float  v8f;

#define B_SZ     8192
#define H1_SZ    128
#define H2_SZ    64
#define PAIRS_SZ 1024
#define OUT_SZ   8192

// f32 -> bf16 round-to-nearest-even (prep kernels only).
static __device__ __forceinline__ bf16_t f2bf(float f) {
  unsigned u = __builtin_bit_cast(unsigned, f);
  u += 0x7FFFu + ((u >> 16) & 1u);
  unsigned short h = (unsigned short)(u >> 16);
  return __builtin_bit_cast(bf16_t, h);
}

// store float's top 16 bits as bf16 (truncation) -> ds_store_b16_d16_hi
static __device__ __forceinline__ void store_bf16_hi(unsigned short* p, float f) {
  *p = (unsigned short)(__builtin_bit_cast(unsigned, f) >> 16);
}

static __device__ __forceinline__ float bf16bits_to_f32(unsigned short h) {
  return __builtin_bit_cast(float, ((unsigned)h) << 16);
}

// single-instruction relu: v_med3_f32(x, 0, +MAX) (no canonicalize double-max)
static __device__ __forceinline__ float relu1(float x) {
  return __builtin_amdgcn_fmed3f(x, 0.0f, 3.4028234663852886e38f);
}

static __device__ __forceinline__ v8f wmma_bf16(v16bf a, v16bf b, v8f c) {
  return __builtin_amdgcn_wmma_f32_16x16x32_bf16(false, a, false, b, (short)0,
                                                 c, false, false);
}

static __device__ __forceinline__ v16bf cat8(v8bf lo, v8bf hi) {
  return __builtin_shufflevector(lo, hi, 0, 1, 2, 3, 4, 5, 6, 7,
                                 8, 9, 10, 11, 12, 13, 14, 15);
}

// ---------------------------------------------------------------------------
// Prep kernels: pack inputs/weights into bf16 with WMMA-friendly layouts.
// ---------------------------------------------------------------------------

__global__ void prep_xzpk(const float* __restrict__ x,
                          const float* __restrict__ z,
                          bf16_t* __restrict__ xzpk) {
  int idx = blockIdx.x * blockDim.x + threadIdx.x;
  if (idx >= B_SZ * 64) return;
  int b = idx >> 6, k = idx & 63;
  float v = (k < 32) ? x[b * 32 + k] : z[b * 32 + (k - 32)];
  xzpk[idx] = f2bf(v);
}

// xzwT[n][k] = bf16(xzw[k][n])  (column-major B-operand layout)
__global__ void prep_xzwT(const float* __restrict__ xzw,
                          bf16_t* __restrict__ xzwT) {
  int idx = blockIdx.x * blockDim.x + threadIdx.x;
  if (idx >= OUT_SZ * 64) return;
  int n = idx >> 6, k = idx & 63;
  xzwT[idx] = f2bf(xzw[(size_t)k * OUT_SZ + n]);
}

// pw1pk[p][row][j], row in [0,32), j in [0,16):
//   row<16 (= hidden unit h): j<8 -> pw1[p][j][h]; else 0
//   row>=16 (= h+16): j==0 -> pb1[p][h] (K=8 bias row); else 0
__global__ void prep_pw1pk(const float* __restrict__ pw1,
                           const float* __restrict__ pb1,
                           bf16_t* __restrict__ pw1pk) {
  int idx = blockIdx.x * blockDim.x + threadIdx.x;
  if (idx >= PAIRS_SZ * 32 * 16) return;
  int p = idx >> 9, rem = idx & 511;
  int row = rem >> 4, j = rem & 15;
  float val = 0.0f;
  if (row < 16) {
    if (j < 8)       val = pw1[p * 128 + j * 16 + row];
  } else if (j == 0) val = pb1[p * 16 + row - 16];
  pw1pk[idx] = f2bf(val);
}

// Column-major bf16 transposes of the main-MLP weights.
__global__ void prep_w(const float* __restrict__ xw1, const float* __restrict__ zw1,
                       const float* __restrict__ xw2, const float* __restrict__ zw2,
                       bf16_t* __restrict__ xw1T, bf16_t* __restrict__ zw1T,
                       bf16_t* __restrict__ xw2T, bf16_t* __restrict__ zw2T) {
  int idx = blockIdx.x * blockDim.x + threadIdx.x;
  const int N1 = H1_SZ * 32;    // 4096 per matrix
  const int N2 = H2_SZ * H1_SZ; // 8192 per matrix
  if (idx < 2 * N1) {
    int sel = idx >= N1;
    int r = idx - sel * N1;
    int n = r >> 5, k = r & 31;
    const float* src = sel ? zw1 : xw1;
    bf16_t* dst = sel ? zw1T : xw1T;
    dst[n * 32 + k] = f2bf(src[k * H1_SZ + n]);
  } else {
    int j = idx - 2 * N1;
    if (j >= 2 * N2) return;
    int sel = j >= N2;
    int r = j - sel * N2;
    int n = r >> 7, k = r & 127;
    const float* src = sel ? zw2 : xw2;
    bf16_t* dst = sel ? zw2T : xw2T;
    dst[n * 128 + k] = f2bf(src[k * H2_SZ + n]);
  }
}

// ---------------------------------------------------------------------------
// Fused interaction + per-pair-MLP kernel.
// Block = 16 batch rows, 8 waves; wave w owns columns [w*1024,(w+1)*1024)
// in 64 tiles of 16 (= 2 pairs). Per tile:
//   2 WMMA (K=64 interaction GEMM, C preloaded with xzb bias) -> med3 relu ->
//   bf16 d16_hi stores to LDS [m][n] ->
//   per pair: 1 ds_load_b128 A (lanes 16-31 hit a constant 1.0 row) +
//             1 32B global load B (rows 16-31 hold pb1 at j=0, else 0) ->
//   WMMA with inline-0 C (bias rides at K=8) -> relu * pw2 accumulate.
// End: shfl butterfly over h, LDS cross-wave sum. No atomics: deterministic.
// ---------------------------------------------------------------------------
__global__ __launch_bounds__(256) void interact_kernel(
    const bf16_t* __restrict__ xzpk, const bf16_t* __restrict__ xzwT,
    const bf16_t* __restrict__ pw1pk, const float* __restrict__ xzb,
    const float* __restrict__ pw2, float* __restrict__ out) {
  __shared__ unsigned short ldsB[8][16][16];  // per-wave h tile, bf16, [m][n]
  __shared__ unsigned short ldsK[8];          // {1.0bf16, 0...}: A bias row
  __shared__ float ldsAcc[8][16];

  int tid = threadIdx.x;
  int wv = tid >> 5;
  int lane = tid & 31;
  int lr = lane & 15;   // column / N index within tile
  int lh = lane >> 4;   // lane half (selects K sub-range)
  int m0 = blockIdx.x * 16;

  if (tid < 8) ldsK[tid] = (tid == 0) ? (unsigned short)0x3F80 : (unsigned short)0;
  __syncthreads();

  // A operands (xz rows, reused for all 64 tiles of this wave).
  const bf16_t* arow = xzpk + (size_t)(m0 + lr) * 64;
  int kb = lh * 16;
  v16bf A0 = cat8(*(const v8bf*)(arow + kb), *(const v8bf*)(arow + kb + 8));
  v16bf A1 = cat8(*(const v8bf*)(arow + 32 + kb),
                  *(const v8bf*)(arow + 32 + kb + 8));

  // Hoisted per-lane LDS addresses for the per-pair A operand:
  // lanes 0-15 read their h row; lanes 16-31 read the constant 1.0 row
  // (their halves map to K=8..15 where only K=8 has nonzero B).
  const unsigned short* aAddr0 = lh ? &ldsK[0] : &ldsB[wv][lr][0];
  const unsigned short* aAddr1 = lh ? &ldsK[0] : &ldsB[wv][lr][8];
  // Hoisted per-lane row offset into pw1pk: lanes 16-31 fetch the pb1 bias rows.
  const size_t bRowOff = (size_t)(lr + lh * 16) * 16;

  const v8bf zero8 = {};
  v8f acc = {};  // acc[v] ~ partial out_xz for (m=v+8*lh) at hidden h=lr

  for (int t = 0; t < 64; ++t) {
    int n0 = wv * 1024 + t * 16;

    // Interaction GEMM B operand (column-major xzwT).
    const bf16_t* bcol = xzwT + (size_t)(n0 + lr) * 64 + lh * 16;
    v16bf B0 = cat8(*(const v8bf*)(bcol), *(const v8bf*)(bcol + 8));
    v16bf B1 = cat8(*(const v8bf*)(bcol + 32), *(const v8bf*)(bcol + 40));

    // xzb bias rides in the C operand (same for all 8 rows of this column).
    float bias = xzb[n0 + lr];
    v8f c;
#pragma unroll
    for (int v = 0; v < 8; ++v) c[v] = bias;
    c = wmma_bf16(A0, B0, c);   // K = 0..31
    c = wmma_bf16(A1, B1, c);   // K = 32..63

    // relu (1 op), store bf16 (= f32 top half) into LDS transpose tile.
#pragma unroll
    for (int v = 0; v < 8; ++v)
      store_bf16_hi(&ldsB[wv][v + 8 * lh][lr], relu1(c[v]));
    asm volatile("s_wait_dscnt 0" ::: "memory");

    // Two pairs per tile: tiny MLP layer1 as WMMA, bias at K=8, C = inline 0.
#pragma unroll
    for (int pl = 0; pl < 2; ++pl) {
      int p = (n0 >> 3) + pl;
      v8bf alo = *(const v8bf*)(pl ? aAddr1 : aAddr0);
      const bf16_t* brow = pw1pk + (size_t)p * 512 + bRowOff;
      v16bf A2 = cat8(alo, zero8);  // upper K rows are don't-care (B==0)
      v16bf B2 = cat8(*(const v8bf*)brow, *(const v8bf*)(brow + 8));

      v8f c2 = {};
      c2 = wmma_bf16(A2, B2, c2);   // D = h@w1 + pb1 (bias row at K=8)

      float w2 = pw2[p * 16 + lr];
#pragma unroll
      for (int v = 0; v < 8; ++v)
        acc[v] += relu1(c2[v]) * w2;
    }
  }

  // Sum over hidden index h (= lr) within each 16-lane half.
#pragma unroll
  for (int v = 0; v < 8; ++v) {
    float s = acc[v];
    s += __shfl_xor(s, 1, 32);
    s += __shfl_xor(s, 2, 32);
    s += __shfl_xor(s, 4, 32);
    s += __shfl_xor(s, 8, 32);
    acc[v] = s;
  }
  if (lr == 0) {
#pragma unroll
    for (int v = 0; v < 8; ++v)
      ldsAcc[wv][lh * 8 + v] = acc[v];
  }
  __syncthreads();
  if (tid < 16) {
    float s = 0.0f;
#pragma unroll
    for (int w = 0; w < 8; ++w) s += ldsAcc[w][tid];  // fixed order
    out[m0 + tid] = s;  // out_xz; main-effect MLPs added by next kernel
  }
}

// ---------------------------------------------------------------------------
// Main-effect MLPs (x and z paths), WMMA per 16-row batch tile.
// Block = 64 threads: wave0 = X path, wave1 = Z path. Adds into d_out.
// ---------------------------------------------------------------------------
__global__ __launch_bounds__(64) void mlp_kernel(
    const bf16_t* __restrict__ xzpk,
    const bf16_t* __restrict__ xw1T, const bf16_t* __restrict__ zw1T,
    const bf16_t* __restrict__ xw2T, const bf16_t* __restrict__ zw2T,
    const float* __restrict__ xb1, const float* __restrict__ zb1,
    const float* __restrict__ xb2, const float* __restrict__ zb2,
    const float* __restrict__ xw3, const float* __restrict__ zw3,
    const float* __restrict__ xb3, const float* __restrict__ zb3,
    float* __restrict__ out) {
  __shared__ unsigned short h1[2][16][128];  // 8 KB, bf16
  __shared__ unsigned short h2[2][16][64];   // 4 KB, bf16
  __shared__ float res[2][16];

  int tid = threadIdx.x;
  int wv = tid >> 5;
  int lane = tid & 31;
  int lr = lane & 15;
  int lh = lane >> 4;
  int m0 = blockIdx.x * 16;

  const bf16_t* w1T = wv ? zw1T : xw1T;
  const bf16_t* w2T = wv ? zw2T : xw2T;
  const float* b1 = wv ? zb1 : xb1;
  const float* b2 = wv ? zb2 : xb2;
  const float* w3 = wv ? zw3 : xw3;
  float b3 = (wv ? zb3 : xb3)[0];

  const bf16_t* arow = xzpk + (size_t)(m0 + lr) * 64 + wv * 32;
  int kb = lh * 16;
  v16bf A = cat8(*(const v8bf*)(arow + kb), *(const v8bf*)(arow + kb + 8));

  // Layer 1: [16,32] @ [32,128]; bias in C, 8 column tiles, 1 WMMA each.
  for (int nt = 0; nt < 8; ++nt) {
    int n0 = nt * 16;
    const bf16_t* bcol = w1T + (size_t)(n0 + lr) * 32 + lh * 16;
    v16bf Bm = cat8(*(const v8bf*)bcol, *(const v8bf*)(bcol + 8));
    float bias = b1[n0 + lr];
    v8f c;
#pragma unroll
    for (int v = 0; v < 8; ++v) c[v] = bias;
    c = wmma_bf16(A, Bm, c);
#pragma unroll
    for (int v = 0; v < 8; ++v)
      store_bf16_hi(&h1[wv][v + 8 * lh][n0 + lr], relu1(c[v]));
  }
  asm volatile("s_wait_dscnt 0" ::: "memory");

  // Layer 2: [16,128] @ [128,64]; 4 bf16 A chunks straight from LDS.
  v16bf A2[4];
#pragma unroll
  for (int kcI = 0; kcI < 4; ++kcI) {
    int kc = kcI * 32;
    A2[kcI] = cat8(*(const v8bf*)(&h1[wv][lr][kc + kb]),
                   *(const v8bf*)(&h1[wv][lr][kc + kb + 8]));
  }
  for (int nt = 0; nt < 4; ++nt) {
    int n0 = nt * 16;
    float bias = b2[n0 + lr];
    v8f c;
#pragma unroll
    for (int v = 0; v < 8; ++v) c[v] = bias;
#pragma unroll
    for (int kcI = 0; kcI < 4; ++kcI) {
      const bf16_t* bcol = w2T + (size_t)(n0 + lr) * 128 + kcI * 32 + lh * 16;
      v16bf Bm = cat8(*(const v8bf*)bcol, *(const v8bf*)(bcol + 8));
      c = wmma_bf16(A2[kcI], Bm, c);
    }
#pragma unroll
    for (int v = 0; v < 8; ++v)
      store_bf16_hi(&h2[wv][v + 8 * lh][n0 + lr], relu1(c[v]));
  }
  asm volatile("s_wait_dscnt 0" ::: "memory");

  // Layer 3: dot with w3[64]; each half-lane sums 32 k's, shfl_xor(16) joins.
  float s = 0.0f;
#pragma unroll
  for (int k = 0; k < 32; ++k)
    s += bf16bits_to_f32(h2[wv][lr][lh * 32 + k]) * w3[lh * 32 + k];
  s += __shfl_xor(s, 16, 32);
  if (lane < 16) res[wv][lane] = s + b3;
  __syncthreads();
  if (tid < 16) out[m0 + tid] += res[0][tid] + res[1][tid];
}

// ---------------------------------------------------------------------------
// Launch: prep (bf16 pack/transpose, ~3.2MB ws) -> fused interaction ->
// main-effect MLPs (accumulate). All on `stream`; no atomics; deterministic.
// ---------------------------------------------------------------------------
extern "C" void kernel_launch(void* const* d_in, const int* in_sizes, int n_in,
                              void* d_out, int out_size, void* d_ws, size_t ws_size,
                              hipStream_t stream) {
  (void)in_sizes; (void)n_in; (void)out_size; (void)ws_size;
  const float* x   = (const float*)d_in[0];
  const float* z   = (const float*)d_in[1];
  const float* xw1 = (const float*)d_in[2];
  const float* xb1 = (const float*)d_in[3];
  const float* xw2 = (const float*)d_in[4];
  const float* xb2 = (const float*)d_in[5];
  const float* xw3 = (const float*)d_in[6];
  const float* xb3 = (const float*)d_in[7];
  const float* zw1 = (const float*)d_in[8];
  const float* zb1 = (const float*)d_in[9];
  const float* zw2 = (const float*)d_in[10];
  const float* zb2 = (const float*)d_in[11];
  const float* zw3 = (const float*)d_in[12];
  const float* zb3 = (const float*)d_in[13];
  const float* xzw = (const float*)d_in[14];
  const float* xzb = (const float*)d_in[15];
  const float* pw1 = (const float*)d_in[16];
  const float* pb1 = (const float*)d_in[17];
  const float* pw2 = (const float*)d_in[18];
  float* out = (float*)d_out;

  char* w = (char*)d_ws;
  bf16_t* xzpk  = (bf16_t*)w; w += (size_t)B_SZ * 64 * 2;            // 1 MB
  bf16_t* xzwT  = (bf16_t*)w; w += (size_t)OUT_SZ * 64 * 2;          // 1 MB
  bf16_t* pw1pk = (bf16_t*)w; w += (size_t)PAIRS_SZ * 32 * 16 * 2;   // 1 MB
  bf16_t* xw1T  = (bf16_t*)w; w += 128 * 32 * 2;
  bf16_t* zw1T  = (bf16_t*)w; w += 128 * 32 * 2;
  bf16_t* xw2T  = (bf16_t*)w; w += 64 * 128 * 2;
  bf16_t* zw2T  = (bf16_t*)w; w += 64 * 128 * 2;

  prep_xzpk<<<(B_SZ * 64 + 255) / 256, 256, 0, stream>>>(x, z, xzpk);
  prep_xzwT<<<(OUT_SZ * 64 + 255) / 256, 256, 0, stream>>>(xzw, xzwT);
  prep_pw1pk<<<(PAIRS_SZ * 32 * 16 + 255) / 256, 256, 0, stream>>>(pw1, pb1,
                                                                   pw1pk);
  prep_w<<<(2 * 128 * 32 + 2 * 64 * 128 + 255) / 256, 256, 0, stream>>>(
      xw1, zw1, xw2, zw2, xw1T, zw1T, xw2T, zw2T);

  interact_kernel<<<B_SZ / 16, 256, 0, stream>>>(xzpk, xzwT, pw1pk, xzb, pw2,
                                                 out);
  mlp_kernel<<<B_SZ / 16, 64, 0, stream>>>(xzpk, xw1T, zw1T, xw2T, zw2T,
                                           xb1, zb1, xb2, zb2, xw3, zw3,
                                           xb3, zb3, out);
}
